// Reservoir_12936441495810
// MI455X (gfx1250) — matrix operand order repaired and checked by reading, not compiled, not run on hardware
//
#include <hip/hip_runtime.h>
#include <hip/hip_bf16.h>
#include <math.h>

// ---------------------------------------------------------------------------
// 2-layer leaky ESN on MI455X (gfx1250).
//   pre = x_t @ W_in.T + h @ W_hat.T ;  h' = 0.5*h + 0.5*tanh(pre)
// 32 persistent WGs = 2 batch-groups (32 rows) x 16 H-slices (64 cols).
// Weight slice (bf16) resident in LDS (64 x 2048 + pad = 257KB of the 320KB
// CDNA5 WGP LDS). GEMM: v_wmma_f32_16x16x32_bf16, fp32 accumulate.
// fp32 master state lives in d_out; a double-buffered bf16 mirror of the
// state ([2][64][1024], 256KB, L2-resident) feeds the h-part A-fragments
// directly in bf16 (no converts in the hot loop).
// Per-step sync: 16-WG flag barrier within each batch group only.
// Epilogue tanh: branchless 0.5 - rcp(exp(2x)+1) (v_exp_f32 + v_rcp_f32).
// ---------------------------------------------------------------------------

typedef __attribute__((ext_vector_type(16))) __bf16 v16bf;
typedef __attribute__((ext_vector_type(8)))  __bf16 v8bf;
typedef __attribute__((ext_vector_type(8)))  float  v8f;

#define T_STEPS  512
#define BATCH    64
#define IN_DIM   128
#define H_DIM    1024
#define N_SLICE  16          // H slices (one WG each)
#define HS       64          // H columns per slice
#define BT       32          // batch rows per group
#define LDSK     2056        // padded LDS row stride in bf16 elems (2048 + 8)
#define NTHREADS 256

__global__ void init_flags_kernel(int* flags) {
    if (threadIdx.x < 32) flags[threadIdx.x] = 0;
}

__device__ __forceinline__ void group_barrier(int* flags, int g, int s,
                                              int target, int tid) {
    __threadfence();                         // release my slice (agent scope)
    __syncthreads();
    if (tid == 0)
        __hip_atomic_store(&flags[g * N_SLICE + s], target,
                           __ATOMIC_RELAXED, __HIP_MEMORY_SCOPE_AGENT);
    if (tid < N_SLICE) {
        while (__hip_atomic_load(&flags[g * N_SLICE + tid],
                                 __ATOMIC_RELAXED,
                                 __HIP_MEMORY_SCOPE_AGENT) < target)
            __builtin_amdgcn_s_sleep(1);
    }
    __syncthreads();
    // acquire: invalidate WGP-local cache so peer slices are fresh
    __builtin_amdgcn_fence(__ATOMIC_ACQUIRE, "agent");
}

// 0.5*tanh(p) computed branchlessly: 0.5 - rcp(exp(2p) + 1)
__device__ __forceinline__ float half_tanh(float p) {
    const float e = __expf(2.0f * p);        // v_exp_f32 path, saturates clean
    return 0.5f - __builtin_amdgcn_rcpf(e + 1.0f);
}

// bf16 A-fragment from 16 fp32 values (two 8-elem chunks, converted)
__device__ __forceinline__ v16bf afrag_f32(const float* lo, const float* hif) {
    const v8f f0 = *(const v8f*)lo;
    const v8f f1 = *(const v8f*)hif;
    v16bf a;
#pragma unroll
    for (int i = 0; i < 8; ++i) {
        a[i]     = (__bf16)f0[i];
        a[i + 8] = (__bf16)f1[i];
    }
    return a;
}

// bf16 A-fragment from two already-bf16 8-elem chunks (pure register concat)
__device__ __forceinline__ v16bf afrag_bf16(const __bf16* lo,
                                            const __bf16* hif) {
    const v8bf c0 = *(const v8bf*)lo;
    const v8bf c1 = *(const v8bf*)hif;
    v16bf a;
#pragma unroll
    for (int i = 0; i < 8; ++i) {
        a[i]     = c0[i];
        a[i + 8] = c1[i];
    }
    return a;
}

template <int LAYER>
__device__ __forceinline__ void run_layer(
    const float* __restrict__ xin,    // [T][B][XSTR] input trajectory base
    const float* __restrict__ Win,
    const float* __restrict__ What,
    float* __restrict__ out,          // [T][B][2][H] fp32
    __bf16* __restrict__ hbf,         // [2][B][H] bf16 state mirror
    int*    __restrict__ flags,
    __bf16* __restrict__ lds_w,
    int tid)
{
    constexpr int KIN  = LAYER ? H_DIM : IN_DIM;
    constexpr int XSTR = LAYER ? (2 * H_DIM) : IN_DIM;

    const int s    = blockIdx.x & 15;
    const int g    = blockIdx.x >> 4;
    const int wave = tid >> 5;
    const int lane = tid & 31;
    const int r    = wave & 1;
    const int c    = wave >> 1;
    const int m    = lane & 15;
    const int hi   = lane >> 4;
    const int bR   = g * BT + r * 16;
    const int jloc = c * 16;
    const int jglb = s * HS + jloc;
    const int bRow = bR + m;

    // ---- stage weight slice into LDS as bf16, once per layer --------------
    {
        constexpr int KTOT = KIN + H_DIM;
        constexpr int KQ   = KTOT >> 2;
        for (int q = tid; q < HS * KQ; q += NTHREADS) {
            const int row = q / KQ;
            const int col = (q - row * KQ) << 2;
            const float* src = (col < KIN)
                ? (Win  + (size_t)(s * HS + row) * KIN   + col)
                : (What + (size_t)(s * HS + row) * H_DIM + (col - KIN));
            const float4 v = *(const float4*)src;
            __bf16* dst = &lds_w[row * LDSK + col];
            dst[0] = (__bf16)v.x; dst[1] = (__bf16)v.y;
            dst[2] = (__bf16)v.z; dst[3] = (__bf16)v.w;
        }
    }
    __syncthreads();

    for (int t = 0; t < T_STEPS; ++t) {
        v8f acc = {0.f, 0.f, 0.f, 0.f, 0.f, 0.f, 0.f, 0.f};

        // ---- x-part GEMM: A = x_t (fp32 -> bf16 in regs) ------------------
        {
            const float* arow = xin + (size_t)t * BATCH * XSTR
                                    + (size_t)bRow * XSTR + 8 * hi;
            const __bf16* wrow = &lds_w[(jloc + m) * LDSK + 8 * hi];
#pragma unroll 2
            for (int k0 = 0; k0 < KIN; k0 += 32) {
                const v16bf a = afrag_f32(arow + k0, arow + k0 + 16);
                const v16bf b = afrag_bf16(wrow + k0, wrow + k0 + 16);
                acc = __builtin_amdgcn_wmma_f32_16x16x32_bf16(
                          false, a, false, b, (short)0, acc, false, false);
            }
        }

        // ---- h-part GEMM: A = h(t-1) from bf16 mirror (no converts) -------
        if (t > 0) {
            const __bf16* hrow = hbf
                + (size_t)((t - 1) & 1) * BATCH * H_DIM
                + (size_t)bRow * H_DIM + 8 * hi;
            const __bf16* wrow = &lds_w[(jloc + m) * LDSK + KIN + 8 * hi];
#pragma unroll 2
            for (int k0 = 0; k0 < H_DIM; k0 += 32) {
                const v16bf a = afrag_bf16(hrow + k0, hrow + k0 + 16);
                const v16bf b = afrag_bf16(wrow + k0, wrow + k0 + 16);
                acc = __builtin_amdgcn_wmma_f32_16x16x32_bf16(
                          false, a, false, b, (short)0, acc, false, false);
            }
        }

        // ---- leaky update + store (C layout: VGPR i -> M=i+8*hi, N=m) -----
        {
            const size_t colOff = (size_t)LAYER * H_DIM + (size_t)(jglb + m);
            const size_t tBase  = (size_t)t * BATCH * (2 * H_DIM);
            __bf16* hdst = hbf + (size_t)(t & 1) * BATCH * H_DIM;
#pragma unroll
            for (int i = 0; i < 8; ++i) {
                const int    brow = bR + i + 8 * hi;
                const size_t oidx = tBase + (size_t)brow * (2 * H_DIM) + colOff;
                float hnew;
                if (t == 0) {
                    hnew = half_tanh(acc[i]);
                } else {
                    const float hold = out[oidx - (size_t)BATCH * (2 * H_DIM)];
                    hnew = 0.5f * hold + half_tanh(acc[i]);
                }
                out[oidx] = hnew;
                hdst[(size_t)brow * H_DIM + (jglb + m)] = (__bf16)hnew;
            }
        }

        // prefetch next step's x rows into cache while we sync
        if (t + 1 < T_STEPS) {
            const float* nx = xin + (size_t)(t + 1) * BATCH * XSTR
                                  + (size_t)bRow * XSTR + 8 * hi;
            __builtin_prefetch(nx, 0, 3);        // global_prefetch_b8
        }

        group_barrier(flags, g, s, LAYER * T_STEPS + t + 1, tid);
    }
    __syncthreads();                             // before LDS re-stage
}

__global__ void __launch_bounds__(NTHREADS, 1)
reservoir_persistent(const float* __restrict__ x,
                     const float* __restrict__ Win0,
                     const float* __restrict__ What0,
                     const float* __restrict__ Win1,
                     const float* __restrict__ What1,
                     float* __restrict__ out,
                     int* __restrict__ flags,
                     __bf16* __restrict__ hbf)
{
    extern __shared__ __bf16 lds_w[];            // [HS][LDSK]
    const int tid = threadIdx.x;

    run_layer<0>(x,   Win0, What0, out, hbf, flags, lds_w, tid);
    run_layer<1>(out, Win1, What1, out, hbf, flags, lds_w, tid);
}

extern "C" void kernel_launch(void* const* d_in, const int* in_sizes, int n_in,
                              void* d_out, int out_size, void* d_ws,
                              size_t ws_size, hipStream_t stream) {
    const float* x     = (const float*)d_in[0];
    const float* Win0  = (const float*)d_in[1];
    const float* What0 = (const float*)d_in[2];
    const float* Win1  = (const float*)d_in[3];
    const float* What1 = (const float*)d_in[4];
    float*  out   = (float*)d_out;
    int*    flags = (int*)d_ws;                          // 32 ints
    __bf16* hbf   = (__bf16*)((char*)d_ws + 256);        // [2][64][1024] bf16

    const int ldsBytes = HS * LDSK * 2;          // 263,168 B < 320 KB
    (void)hipFuncSetAttribute((const void*)reservoir_persistent,
                              hipFuncAttributeMaxDynamicSharedMemorySize,
                              ldsBytes);

    init_flags_kernel<<<1, 64, 0, stream>>>(flags);
    reservoir_persistent<<<dim3(32), dim3(NTHREADS), ldsBytes, stream>>>(
        x, Win0, What0, Win1, What1, out, flags, hbf);
}
